// Discrimator_4612794876145
// MI455X (gfx1250) — matrix operand order repaired and checked by reading, not compiled, not run on hardware
//
#include <hip/hip_runtime.h>
#include <hip/hip_bf16.h>

typedef __attribute__((ext_vector_type(16))) __bf16 v16bf;
typedef __attribute__((ext_vector_type(8)))  float  v8f;

#define D 128
#define NGRAPH 8

// ---------------------------------------------------------------- zero
__global__ void zero_f4_kernel(float4* __restrict__ p, int n4) {
  int i = blockIdx.x * blockDim.x + threadIdx.x;
  if (i < n4) p[i] = make_float4(0.f, 0.f, 0.f, 0.f);
}

// -------------------------------------------- W -> split-bf16, transposed
// Produces WhT/WlT in [n][k] layout so a WMMA B-fragment (fixed n, 16
// consecutive k) is one contiguous 32B load per lane.
__global__ void convw_kernel(const float* __restrict__ W,
                             __bf16* __restrict__ hT, __bf16* __restrict__ lT) {
  int i = blockIdx.x * blockDim.x + threadIdx.x;   // over D*D
  if (i < D * D) {
    int n = i & (D - 1);
    int k = i >> 7;
    float w = W[k * D + n];
    __bf16 hi = (__bf16)w;
    hT[n * D + k] = hi;
    lT[n * D + k] = (__bf16)(w - (float)hi);
  }
}

// ---------------------------------------------------------------- degrees
__global__ void degree_kernel(const int* __restrict__ src, const int* __restrict__ dst,
                              float* __restrict__ deg_out, float* __restrict__ deg_in, int nE) {
  int i = blockIdx.x * blockDim.x + threadIdx.x;
  if (i < nE) {
    atomicAdd(&deg_out[src[i]], 1.f);
    atomicAdd(&deg_in[dst[i]], 1.f);
  }
}

__global__ void rs_kernel(float* __restrict__ p, int n) {
  int i = blockIdx.x * blockDim.x + threadIdx.x;
  if (i < n) p[i] = rsqrtf(fmaxf(p[i], 1.f));
}

// ------------------------------------------------- GEMM: out = x @ W (WMMA)
// Split-bf16: x = xh + xl, W = Wh + Wl; acc += xh*Wh + xh*Wl + xl*Wh.
// Block = 256 threads = 8 waves; block covers 16 rows, wave w covers cols 16w..16w+15.
__global__ __launch_bounds__(256)
void gemm_wmma_kernel(const float* __restrict__ x, const __bf16* __restrict__ WhT,
                      const __bf16* __restrict__ WlT, float* __restrict__ out, int nNodes) {
  const int lane = threadIdx.x & 31;
  const int wave = threadIdx.x >> 5;     // column tile
  const int half = lane >> 4;            // 0: lanes 0-15, 1: lanes 16-31
  const int ml   = lane & 15;
  const int row0 = blockIdx.x * 16;
  const int n0   = wave * 16;

  int row = row0 + ml;
  if (row >= nNodes) row = nNodes - 1;   // clamp loads on last (partial) tile
  const float* xr = x + (size_t)row * D;
  const int n = n0 + ml;                 // B column carried by this lane

  v8f acc = {0.f, 0.f, 0.f, 0.f, 0.f, 0.f, 0.f, 0.f};

#pragma unroll
  for (int kb = 0; kb < D; kb += 32) {
    // A 16x32 as loaded here (contiguous k, same order as B): lane(half,m) elems 0..7 -> K=kb+16*half+j,
    // elems 8..15 -> K=kb+16*half+8+j.
    const float* p0 = xr + kb + 16 * half;
    const float* p1 = xr + kb + 16 * half + 8;
    float af[16];
    { float4 t = *(const float4*)(p0);     af[0]=t.x; af[1]=t.y; af[2]=t.z; af[3]=t.w; }
    { float4 t = *(const float4*)(p0 + 4); af[4]=t.x; af[5]=t.y; af[6]=t.z; af[7]=t.w; }
    { float4 t = *(const float4*)(p1);     af[8]=t.x; af[9]=t.y; af[10]=t.z; af[11]=t.w; }
    { float4 t = *(const float4*)(p1 + 4); af[12]=t.x; af[13]=t.y; af[14]=t.z; af[15]=t.w; }
    v16bf ah, al;
#pragma unroll
    for (int i = 0; i < 16; ++i) {
      __bf16 hi = (__bf16)af[i];
      ah[i] = hi;
      al[i] = (__bf16)(af[i] - (float)hi);
    }
    // B 32x16: lane n, elems j=0..15 -> K = kb + 16*half + j (contiguous 32B).
    const v16bf bh = *(const v16bf*)(WhT + (size_t)n * D + kb + 16 * half);
    const v16bf bl = *(const v16bf*)(WlT + (size_t)n * D + kb + 16 * half);

    acc = __builtin_amdgcn_wmma_f32_16x16x32_bf16(false, ah, false, bh, (short)0, acc, false, false);
    acc = __builtin_amdgcn_wmma_f32_16x16x32_bf16(false, ah, false, bl, (short)0, acc, false, false);
    acc = __builtin_amdgcn_wmma_f32_16x16x32_bf16(false, al, false, bh, (short)0, acc, false, false);
  }

  // C/D layout: VGPR r -> row = row0 + r + 8*half, col = n0 + ml.
#pragma unroll
  for (int r = 0; r < 8; ++r) {
    int orow = row0 + r + 8 * half;
    if (orow < nNodes) out[(size_t)orow * D + n0 + ml] = acc[r];
  }
}

// ------------------------------------------- edge gather/scale/scatter-add
// One edge per wave; lane handles 4 contiguous floats (coalesced 512B row).
__global__ void scatter_kernel(const float* __restrict__ h, const int* __restrict__ src,
                               const int* __restrict__ dst, const float* __restrict__ rs_out,
                               float* __restrict__ agg, int nE) {
  const int lane = threadIdx.x & 31;
  int e = (blockIdx.x * blockDim.x + threadIdx.x) >> 5;
  const int stride = (gridDim.x * blockDim.x) >> 5;
  for (; e < nE; e += stride) {
    const int s  = src[e];
    const int dd = dst[e];
    const float w = rs_out[s];
    int en = e + stride;
    if (en < nE) {  // prefetch next gather row (global_prefetch_b8)
      __builtin_prefetch(h + (size_t)src[en] * D + lane * 4, 0, 0);
    }
    const float4 v = *(const float4*)(h + (size_t)s * D + lane * 4);
    float* ap = agg + (size_t)dd * D + lane * 4;
    atomicAdd(ap + 0, v.x * w);
    atomicAdd(ap + 1, v.y * w);
    atomicAdd(ap + 2, v.z * w);
    atomicAdd(ap + 3, v.w * w);
  }
}

// ---------------------------------------------- out = relu(agg*rs_in + b)
__global__ void finalize_kernel(const float* __restrict__ agg, const float* __restrict__ rs_in,
                                const float* __restrict__ b, float* __restrict__ out, int total4) {
  int i = blockIdx.x * blockDim.x + threadIdx.x;   // over nNodes * D/4
  if (i < total4) {
    int nn = i >> 5;
    int c4 = (i & 31) * 4;
    float r = rs_in[nn];
    float4 v  = *(const float4*)(agg + (size_t)nn * D + c4);
    float4 bb = *(const float4*)(b + c4);
    float4 o;
    o.x = fmaxf(v.x * r + bb.x, 0.f);
    o.y = fmaxf(v.y * r + bb.y, 0.f);
    o.z = fmaxf(v.z * r + bb.z, 0.f);
    o.w = fmaxf(v.w * r + bb.w, 0.f);
    *(float4*)(out + (size_t)nn * D + c4) = o;
  }
}

// --------------------------------------------------- per-graph mean pooling
__global__ void pool_kernel(const float* __restrict__ h, const int* __restrict__ gid,
                            float* __restrict__ sums, float* __restrict__ cnt, int nNodes) {
  __shared__ float lsum[NGRAPH * D];
  __shared__ float lcnt[NGRAPH];
  for (int i = threadIdx.x; i < NGRAPH * D; i += blockDim.x) lsum[i] = 0.f;
  if (threadIdx.x < NGRAPH) lcnt[threadIdx.x] = 0.f;
  __syncthreads();

  const int lane = threadIdx.x & 31;
  const int wave = threadIdx.x >> 5;
  const int wpb  = blockDim.x >> 5;
  int n = blockIdx.x * wpb + wave;
  const int stride = gridDim.x * wpb;
  for (; n < nNodes; n += stride) {
    const int g = gid[n];
    const float4 v = *(const float4*)(h + (size_t)n * D + lane * 4);
    float* lp = &lsum[g * D + lane * 4];
    atomicAdd(lp + 0, v.x);
    atomicAdd(lp + 1, v.y);
    atomicAdd(lp + 2, v.z);
    atomicAdd(lp + 3, v.w);
    if (lane == 0) atomicAdd(&lcnt[g], 1.f);
  }
  __syncthreads();
  for (int i = threadIdx.x; i < NGRAPH * D; i += blockDim.x) atomicAdd(&sums[i], lsum[i]);
  if (threadIdx.x < NGRAPH) atomicAdd(&cnt[threadIdx.x], lcnt[threadIdx.x]);
}

// ------------------------------------- mean -> PReLU -> Linear -> sigmoid
__global__ void head_kernel(const float* __restrict__ sums, const float* __restrict__ cnt,
                            const float* __restrict__ prelu_a, const float* __restrict__ linW,
                            const float* __restrict__ linb, float* __restrict__ out) {
  const int lane = threadIdx.x & 31;
  const int g    = threadIdx.x >> 5;   // 8 waves, one graph each
  if (g >= NGRAPH) return;
  const float a   = prelu_a[0];
  const float inv = 1.f / fmaxf(cnt[g], 1.f);
  float partial = 0.f;
#pragma unroll
  for (int i = 0; i < 4; ++i) {
    const int dd = lane * 4 + i;
    float p = sums[g * D + dd] * inv;
    p = (p > 0.f) ? p : a * p;
    partial += p * linW[dd];
  }
  for (int off = 16; off > 0; off >>= 1) partial += __shfl_xor(partial, off, 32);
  if (lane == 0) out[g] = 1.f / (1.f + expf(-(partial + linb[0])));
}

// ======================================================================
extern "C" void kernel_launch(void* const* d_in, const int* in_sizes, int n_in,
                              void* d_out, int out_size, void* d_ws, size_t ws_size,
                              hipStream_t stream) {
  (void)n_in; (void)out_size; (void)ws_size;
  const float* x   = (const float*)d_in[0];
  const int*   src = (const int*)d_in[1];
  const int*   dst = (const int*)d_in[2];
  const int*   gid = (const int*)d_in[3];
  const float* W1  = (const float*)d_in[4];
  const float* b1  = (const float*)d_in[5];
  const float* W2  = (const float*)d_in[6];
  const float* b2  = (const float*)d_in[7];
  const float* pa  = (const float*)d_in[8];
  const float* lw  = (const float*)d_in[9];
  const float* lb  = (const float*)d_in[10];
  float* out = (float*)d_out;

  const int nNodes = in_sizes[0] / D;   // 100000
  const int nE     = in_sizes[1];       // 1600000

  // ---- workspace layout (32B-aligned regions) ----
  char* ws = (char*)d_ws;
  const size_t szBuf = (size_t)nNodes * D * sizeof(float);      // 51.2 MB
  float* bufA   = (float*)(ws);
  float* bufB   = (float*)(ws + szBuf);
  float* rs_out = (float*)(ws + 2 * szBuf);                     // deg_out -> rsqrt
  float* rs_in  = rs_out + nNodes;                              // deg_in  -> rsqrt
  float* sums   = rs_in + nNodes;                               // [8][128]
  float* cnt    = sums + NGRAPH * D;                            // [8]
  __bf16* W1hT  = (__bf16*)(cnt + NGRAPH);
  __bf16* W1lT  = W1hT + D * D;
  __bf16* W2hT  = W1lT + D * D;
  __bf16* W2lT  = W2hT + D * D;

  // ---- split + transpose weights (tiny, L2-resident) ----
  convw_kernel<<<(D * D + 255) / 256, 256, 0, stream>>>(W1, W1hT, W1lT);
  convw_kernel<<<(D * D + 255) / 256, 256, 0, stream>>>(W2, W2hT, W2lT);

  // ---- degrees -> rsqrt norms ----
  const int degF4 = (2 * nNodes) / 4;
  zero_f4_kernel<<<(degF4 + 255) / 256, 256, 0, stream>>>((float4*)rs_out, degF4);
  degree_kernel<<<(nE + 255) / 256, 256, 0, stream>>>(src, dst, rs_out, rs_in, nE);
  rs_kernel<<<(2 * nNodes + 255) / 256, 256, 0, stream>>>(rs_out, 2 * nNodes);

  const int nTiles = (nNodes + 15) / 16;          // 6250 exact
  const int bufF4  = (int)(szBuf / 16);
  const int finT4  = nNodes * (D / 4);

  // ---- layer 1 ----
  gemm_wmma_kernel<<<nTiles, 256, 0, stream>>>(x, W1hT, W1lT, bufA, nNodes);
  zero_f4_kernel<<<(bufF4 + 255) / 256, 256, 0, stream>>>((float4*)bufB, bufF4);
  scatter_kernel<<<4096, 256, 0, stream>>>(bufA, src, dst, rs_out, bufB, nE);
  finalize_kernel<<<(finT4 + 255) / 256, 256, 0, stream>>>(bufB, rs_in, b1, bufA, finT4);

  // ---- layer 2 ----
  gemm_wmma_kernel<<<nTiles, 256, 0, stream>>>(bufA, W2hT, W2lT, bufB, nNodes);
  zero_f4_kernel<<<(bufF4 + 255) / 256, 256, 0, stream>>>((float4*)bufA, bufF4);
  scatter_kernel<<<4096, 256, 0, stream>>>(bufB, src, dst, rs_out, bufA, nE);
  finalize_kernel<<<(finT4 + 255) / 256, 256, 0, stream>>>(bufA, rs_in, b2, bufB, finT4);

  // ---- pooling + head ----
  const int poolF4 = (NGRAPH * D + NGRAPH) / 4;   // sums + cnt contiguous, 1032/4
  zero_f4_kernel<<<(poolF4 + 255) / 256, 256, 0, stream>>>((float4*)sums, poolF4);
  pool_kernel<<<512, 256, 0, stream>>>(bufB, gid, sums, cnt, nNodes);
  head_kernel<<<1, 256, 0, stream>>>(sums, cnt, pa, lw, lb, out);
}
